// GAT_54082228191677
// MI455X (gfx1250) — matrix operand order repaired and checked by reading, not compiled, not run on hardware
//
#include <hip/hip_runtime.h>

typedef __attribute__((ext_vector_type(16))) _Float16 v16h;
typedef __attribute__((ext_vector_type(8)))  float    v8f;

#define NEG_SLOPE 0.2f
#define GAT_EPS   1e-16f
#define LDSW      40   // padded LDS row stride in halfs (80B = 5x16B, keeps b128 alignment)

union Frag16 { unsigned int u[8]; v16h h; };

// ---- CDNA5 async global->LDS copy (ASYNCcnt path), 16B per lane ------------
__device__ __forceinline__ unsigned ldsOffset(const _Float16* p) {
    // generic -> LDS(AS3) pointer; ptrtoint of an AS3 pointer is the LDS byte offset
    return (unsigned)(unsigned long long)(const __attribute__((address_space(3))) _Float16*)p;
}
__device__ __forceinline__ void asyncLdsB128(unsigned ldsoff, unsigned long long gaddr) {
    asm volatile("global_load_async_to_lds_b128 %0, %1, off"
                 :: "v"(ldsoff), "v"(gaddr) : "memory");
}
__device__ __forceinline__ void waitAsyncLe4() { asm volatile("s_wait_asynccnt 0x4" ::: "memory"); }
__device__ __forceinline__ void waitAsync0()   { asm volatile("s_wait_asynccnt 0x0" ::: "memory"); }

// ---------------------------------------------------------------------------
// WMMA GEMM: C[N x 256] = A[N x K] * BT^T, A f16 row-major [N][K],
// BT f16 [256][K] (pre-transposed W). 128 threads = 4 waves; block tile 64x64;
// each wave a 32x32 sub-tile as 2x2 v_wmma_f32_16x16x32_f16.
// Double-buffered LDS filled with global_load_async_to_lds_b128.
// ---------------------------------------------------------------------------
__global__ __launch_bounds__(128)
void k_gemm(const _Float16* __restrict__ A, const _Float16* __restrict__ BT,
            float* __restrict__ C, int N, int K) {
    __shared__ _Float16 As[2][64 * LDSW];
    __shared__ _Float16 Bs[2][64 * LDSW];

    const int t       = threadIdx.x;
    const int lane    = t & 31;
    const int wave    = t >> 5;
    const int half    = lane >> 4;   // 0 or 1
    const int l15     = lane & 15;
    const int rowBase = blockIdx.x * 64;
    const int colBase = blockIdx.y * 64;
    const int waveRow = (wave >> 1) * 32;
    const int waveCol = (wave & 1) * 32;

    // two 16B staging slots per matrix per thread: 4 waves x 4 async = 8KB/tile
    int srow[2], sq[2];
    unsigned long long gA[2], gB[2];
    #pragma unroll
    for (int i = 0; i < 2; ++i) {
        int idx = t + i * 128;          // 0..255 b128 slots
        srow[i] = idx >> 2;             // 0..63
        sq[i]   = (idx & 3) * 8;        // halfs: 0,8,16,24
        int gr = rowBase + srow[i]; if (gr >= N) gr = N - 1;
        gA[i] = (unsigned long long)(const void*)(A  + (size_t)gr * K + sq[i]);
        gB[i] = (unsigned long long)(const void*)(BT + (size_t)(colBase + srow[i]) * K + sq[i]);
    }

    v8f acc[2][2];
    acc[0][0] = 0; acc[0][1] = 0; acc[1][0] = 0; acc[1][1] = 0;

    // prologue: tile 0 -> buffer 0
    #pragma unroll
    for (int i = 0; i < 2; ++i) {
        asyncLdsB128(ldsOffset(&As[0][srow[i] * LDSW + sq[i]]), gA[i]);
        asyncLdsB128(ldsOffset(&Bs[0][srow[i] * LDSW + sq[i]]), gB[i]);
    }

    int cur = 0;
    for (int k0 = 0; k0 < K; k0 += 32) {
        const bool more = (k0 + 32) < K;
        if (more) {                       // prefetch next k-tile into other buffer
            const int nxt = cur ^ 1;
            const unsigned long long kb = (unsigned long long)(k0 + 32) * 2;
            #pragma unroll
            for (int i = 0; i < 2; ++i) {
                asyncLdsB128(ldsOffset(&As[nxt][srow[i] * LDSW + sq[i]]), gA[i] + kb);
                asyncLdsB128(ldsOffset(&Bs[nxt][srow[i] * LDSW + sq[i]]), gB[i] + kb);
            }
            waitAsyncLe4();               // oldest 4 (current tile) complete
        } else {
            waitAsync0();
        }
        __syncthreads();

        // fragments per CDNA5 16-bit A(16x32)/B(32x16) VGPR layouts
        Frag16 af[2], bf[2];
        #pragma unroll
        for (int r = 0; r < 2; ++r) {
            int m = waveRow + r * 16 + l15;
            const unsigned int* rowp =
                reinterpret_cast<const unsigned int*>(&As[cur][m * LDSW]);
            #pragma unroll
            for (int j = 0; j < 8; ++j) {
                int kp = ((j < 4) ? (2 * j) : (16 + 2 * (j - 4))) + half * 8;
                af[r].u[j] = rowp[kp >> 1];
            }
        }
        #pragma unroll
        for (int c = 0; c < 2; ++c) {
            int n = waveCol + c * 16 + l15;
            const unsigned int* rowp =
                reinterpret_cast<const unsigned int*>(&Bs[cur][n * LDSW]);
            #pragma unroll
            for (int j = 0; j < 8; ++j) bf[c].u[j] = rowp[(((j < 4) ? (2 * j) : (16 + 2 * (j - 4))) + half * 8) >> 1];
        }
        #pragma unroll
        for (int r = 0; r < 2; ++r)
            #pragma unroll
            for (int c = 0; c < 2; ++c)
                acc[r][c] = __builtin_amdgcn_wmma_f32_16x16x32_f16(
                    false, af[r].h, false, bf[c].h, (short)0, acc[r][c], false, false);
        __syncthreads();                  // done reading buf[cur] before its refill
        cur ^= 1;
    }

    // ---- store; D layout: VGPR v -> M = v + 8*half, N = l15
    if (rowBase + 64 <= N) {              // interior fast path: no guards
        #pragma unroll
        for (int r = 0; r < 2; ++r)
            #pragma unroll
            for (int c = 0; c < 2; ++c)
                #pragma unroll
                for (int v = 0; v < 8; ++v) {
                    int gr = rowBase + waveRow + r * 16 + v + half * 8;
                    C[(size_t)gr * 256 + colBase + waveCol + c * 16 + l15] = acc[r][c][v];
                }
    } else {
        #pragma unroll
        for (int r = 0; r < 2; ++r)
            #pragma unroll
            for (int c = 0; c < 2; ++c)
                #pragma unroll
                for (int v = 0; v < 8; ++v) {
                    int gr = rowBase + waveRow + r * 16 + v + half * 8;
                    if (gr < N)
                        C[(size_t)gr * 256 + colBase + waveCol + c * 16 + l15] = acc[r][c][v];
                }
    }
}

// ---------------------------------------------------------------------------
// one-time precision/layout prep kernels
// ---------------------------------------------------------------------------
__global__ void k_f32_to_f16(const float* __restrict__ in, _Float16* __restrict__ out, long n) {
    long i = (long)blockIdx.x * 256 + threadIdx.x;
    if (i < n) out[i] = (_Float16)in[i];
}

// W [K][256] f32 -> WT [256][K] f16
__global__ void k_w_transpose(const float* __restrict__ W, _Float16* __restrict__ WT, int K) {
    int n = blockIdx.y * 32 + (threadIdx.x & 31);
    int k = blockIdx.x * 8 + (threadIdx.x >> 5);
    WT[(size_t)n * K + k] = (_Float16)W[(size_t)k * 256 + n];
}

// ---------------------------------------------------------------------------
// per-(node,head) attention logits
// ---------------------------------------------------------------------------
__global__ void k_node_att(const float* __restrict__ H, const float* __restrict__ attS,
                           const float* __restrict__ attD, float* __restrict__ aS,
                           float* __restrict__ aD, int NH) {
    int i = blockIdx.x * blockDim.x + threadIdx.x;   // i = n*8 + h
    if (i >= NH) return;
    int h = i & 7;
    const float* hp = H + (size_t)i * 32;
    const float* sp = attS + h * 32;
    const float* dp = attD + h * 32;
    float as = 0.f, ad = 0.f;
    #pragma unroll
    for (int c = 0; c < 32; ++c) { float v = hp[c]; as += v * sp[c]; ad += v * dp[c]; }
    aS[i] = as; aD[i] = ad;
}

__global__ void k_init(float* __restrict__ agg, float* __restrict__ eMax,
                       float* __restrict__ denom, int N) {
    int n = blockIdx.x, t = threadIdx.x;             // block = 256
    agg[(size_t)n * 256 + t] = 0.f;
    if (t < 8) { eMax[n * 8 + t] = -__builtin_inff(); denom[n * 8 + t] = 0.f; }
}

__device__ __forceinline__ void atomicMaxF(float* addr, float v) {
    if (v >= 0.f) atomicMax((int*)addr, __float_as_int(v));
    else          atomicMin((unsigned int*)addr, __float_as_uint(v));
}

__global__ void k_edge_max(const int* __restrict__ src, const int* __restrict__ dst,
                           const float* __restrict__ aS, const float* __restrict__ aD,
                           float* __restrict__ eMax, int E) {
    int e = blockIdx.x * blockDim.x + threadIdx.x;
    if (e >= E) return;
    int s = src[e], d = dst[e];
    #pragma unroll
    for (int h = 0; h < 8; ++h) {
        float v = aS[s * 8 + h] + aD[d * 8 + h];
        v = (v > 0.f) ? v : v * NEG_SLOPE;           // leaky_relu
        atomicMaxF(&eMax[d * 8 + h], v);
    }
}

__global__ void k_edge_p(const int* __restrict__ src, const int* __restrict__ dst,
                         const float* __restrict__ aS, const float* __restrict__ aD,
                         const float* __restrict__ eMax, float* __restrict__ p,
                         float* __restrict__ denom, int E) {
    int e = blockIdx.x * blockDim.x + threadIdx.x;
    if (e >= E) return;
    int s = src[e], d = dst[e];
    #pragma unroll
    for (int h = 0; h < 8; ++h) {
        float v = aS[s * 8 + h] + aD[d * 8 + h];
        v = (v > 0.f) ? v : v * NEG_SLOPE;
        float m = eMax[d * 8 + h];
        if (!(m > -3.0e38f)) m = 0.f;                // isfinite(-inf) -> 0
        float pe = __expf(v - m);
        p[(size_t)e * 8 + h] = pe;
        atomicAdd(&denom[d * 8 + h], pe);
    }
}

// one 256-thread block per edge: agg[dst,h,c] += h[src,h,c] * alpha[e,h]
// (agg is 51MB -> resident in the 192MB L2; atomics stay on-chip)
__global__ __launch_bounds__(256)
void k_edge_msg(const int* __restrict__ src, const int* __restrict__ dst,
                const float* __restrict__ p, const float* __restrict__ denom,
                const float* __restrict__ H, float* __restrict__ agg, int E) {
    int e = blockIdx.x;
    int t = threadIdx.x;
    int h = t >> 5;
    int s = src[e], d = dst[e];
    float alpha = p[(size_t)e * 8 + h] / (denom[d * 8 + h] + GAT_EPS);
    float val = H[(size_t)s * 256 + t] * alpha;
    atomicAdd(&agg[(size_t)d * 256 + t], val);
}

// bias + ELU, emit f16 activations for the next WMMA GEMM
__global__ void k_finalize1(const float* __restrict__ agg, const float* __restrict__ b,
                            _Float16* __restrict__ act, int N) {
    int n = blockIdx.x, t = threadIdx.x;             // block = 256
    size_t i = (size_t)n * 256 + t;
    float v = agg[i] + b[t];
    act[i] = (_Float16)((v > 0.f) ? v : (__expf(v) - 1.f));
}

__global__ void k_finalize2(const float* __restrict__ agg, const float* __restrict__ b,
                            float* __restrict__ out, int N) {
    int n = blockIdx.x, c = threadIdx.x;             // block = 32
    float s = 0.f;
    #pragma unroll
    for (int h = 0; h < 8; ++h) s += agg[(size_t)n * 256 + h * 32 + c];
    out[(size_t)n * 32 + c] = s * 0.125f + b[c];     // mean over heads + bias
}

// ---------------------------------------------------------------------------
extern "C" void kernel_launch(void* const* d_in, const int* in_sizes, int n_in,
                              void* d_out, int out_size, void* d_ws, size_t ws_size,
                              hipStream_t stream) {
    const float* x   = (const float*)d_in[0];
    const int*   ei  = (const int*)d_in[1];
    const float* W1  = (const float*)d_in[2];
    const float* aS1 = (const float*)d_in[3];
    const float* aD1 = (const float*)d_in[4];
    const float* b1  = (const float*)d_in[5];
    const float* W2  = (const float*)d_in[6];
    const float* aS2 = (const float*)d_in[7];
    const float* aD2 = (const float*)d_in[8];
    const float* b2  = (const float*)d_in[9];
    float* out = (float*)d_out;

    const int Fin = 128;
    const int N = in_sizes[0] / Fin;
    const int E = in_sizes[1] / 2;
    const int* src = ei;
    const int* dst = ei + E;

    // ---- workspace layout (f32 region then f16 region, 16B-aligned pieces)
    float* fws   = (float*)d_ws;
    float* bufH  = fws;                              // [N,256] gemm output h
    float* bufA  = bufH  + (size_t)N * 256;          // [N,256] edge aggregator
    float* aS    = bufA  + (size_t)N * 256;          // [N,8]
    float* aD    = aS    + (size_t)N * 8;
    float* eMax  = aD    + (size_t)N * 8;
    float* denom = eMax  + (size_t)N * 8;
    float* pBuf  = denom + (size_t)N * 8;            // [E,8]
    _Float16* x16  = (_Float16*)(pBuf + (size_t)E * 8);  // [N,128]
    _Float16* actH = x16  + (size_t)N * 128;             // [N,256]
    _Float16* WT1  = actH + (size_t)N * 256;             // [256,128]
    _Float16* WT2  = WT1  + 256 * 128;                   // [256,256]

    dim3 gGemm((N + 63) / 64, 4);
    const int NH = N * 8;
    const int eBlocks  = (E + 255) / 256;
    const int nhBlocks = (NH + 255) / 256;

    // ---- one-time f16 conversion / weight transpose (halves GEMM A traffic)
    {
        long nx = (long)N * 128;
        k_f32_to_f16 <<<(unsigned)((nx + 255) / 256), 256, 0, stream>>>(x, x16, nx);
        k_w_transpose<<<dim3(128 / 8, 8), 256, 0, stream>>>(W1, WT1, 128);
        k_w_transpose<<<dim3(256 / 8, 8), 256, 0, stream>>>(W2, WT2, 256);
    }

    // ================= layer 1 (concat) =================
    k_gemm     <<<gGemm,    128, 0, stream>>>(x16, WT1, bufH, N, 128);
    k_node_att <<<nhBlocks, 256, 0, stream>>>(bufH, aS1, aD1, aS, aD, NH);
    k_init     <<<N,        256, 0, stream>>>(bufA, eMax, denom, N);
    k_edge_max <<<eBlocks,  256, 0, stream>>>(src, dst, aS, aD, eMax, E);
    k_edge_p   <<<eBlocks,  256, 0, stream>>>(src, dst, aS, aD, eMax, pBuf, denom, E);
    k_edge_msg <<<E,        256, 0, stream>>>(src, dst, pBuf, denom, bufH, bufA, E);
    k_finalize1<<<N,        256, 0, stream>>>(bufA, b1, actH, N);   // bias+ELU -> f16

    // ================= layer 2 (head mean) ==============
    k_gemm     <<<gGemm,    128, 0, stream>>>(actH, WT2, bufH, N, 256);
    k_node_att <<<nhBlocks, 256, 0, stream>>>(bufH, aS2, aD2, aS, aD, NH);
    k_init     <<<N,        256, 0, stream>>>(bufA, eMax, denom, N);
    k_edge_max <<<eBlocks,  256, 0, stream>>>(src, dst, aS, aD, eMax, E);
    k_edge_p   <<<eBlocks,  256, 0, stream>>>(src, dst, aS, aD, eMax, pBuf, denom, E);
    k_edge_msg <<<E,        256, 0, stream>>>(src, dst, pBuf, denom, bufH, bufA, E);
    k_finalize2<<<N,         32, 0, stream>>>(bufA, b2, out, N);
}
